// KVCacheAttention_25778393710858
// MI455X (gfx1250) — matrix operand order repaired and checked by reading, not compiled, not run on hardware
//
#include <hip/hip_runtime.h>
#include <hip/hip_bf16.h>

// ---------------------------------------------------------------------------
// GQA attention block for MI455X (gfx1250, wave32, WMMA bf16 -> f32 accum)
// Pipeline: bf16 pre-convert -> fused QKV GEMM -> RoPE/relayout ->
//           causal flash attention -> output GEMM
// ---------------------------------------------------------------------------

typedef __attribute__((ext_vector_type(16))) __bf16 bf16x16;
typedef __attribute__((ext_vector_type(8)))  __bf16 bf16x8;
typedef __attribute__((ext_vector_type(8)))  float  f32x8;
typedef __attribute__((ext_vector_type(4)))  int    i32x4;

#define B_DIM   2
#define S_DIM   2048
#define HID     2048
#define NH      16
#define NKVH    4
#define HD      128
#define NQ      (NH * HD)          // 2048
#define NKV     (NKVH * HD)        // 512
#define NTOT    (NQ + 2 * NKV)     // 3072
#define MTOT    (B_DIM * S_DIM)    // 4096
#define SCALE_F 0.08838834764831845f
#define NEG_F   (-1.0e9f)

#if defined(__gfx1250__) && __has_builtin(__builtin_amdgcn_global_load_async_to_lds_b128)
#define HAVE_ASYNC_LDS 1
#else
#define HAVE_ASYNC_LDS 0
#endif

#if HAVE_ASYNC_LDS
// Builtin signature (confirmed by clang diagnostics): param0 = global int4*,
// param1 = LDS int4*, then i32 offset, i32 cpol.
__device__ __forceinline__ void async_copy16(const void* g, void* l) {
  __builtin_amdgcn_global_load_async_to_lds_b128(
      (__attribute__((address_space(1))) i32x4*)(void*)(g),
      (__attribute__((address_space(3))) i32x4*)(l),
      0, 0);
}
__device__ __forceinline__ void async_wait0() {
#if __has_builtin(__builtin_amdgcn_s_wait_asynccnt)
  __builtin_amdgcn_s_wait_asynccnt(0);
#else
  asm volatile("s_wait_asynccnt 0x0" ::: "memory");
#endif
}
#endif

__device__ __forceinline__ f32x8 wmma_bf16(bf16x16 a, bf16x16 b, f32x8 c) {
  // v_wmma_f32_16x16x32_bf16  D = A(16x32) * B(32x16) + C
  return __builtin_amdgcn_wmma_f32_16x16x32_bf16(false, a, false, b,
                                                 (short)0, c, false, false);
}

__device__ __forceinline__ float rowmax16(float v) {
#pragma unroll
  for (int m = 1; m < 16; m <<= 1) v = fmaxf(v, __shfl_xor(v, m, 16));
  return v;
}
__device__ __forceinline__ float rowsum16(float v) {
#pragma unroll
  for (int m = 1; m < 16; m <<= 1) v += __shfl_xor(v, m, 16);
  return v;
}

union AFrag { bf16x16 v; bf16x8 h8[2]; };

// ---------------------------------------------------------------------------
// Kernel 0: fp32 -> bf16 conversion (8 elements / thread, packed cvt).
// ---------------------------------------------------------------------------
__global__ __launch_bounds__(256)
void cvt_bf16_kernel(const float* __restrict__ src, __bf16* __restrict__ dst) {
  const size_t i = ((size_t)blockIdx.x * 256 + threadIdx.x) * 8;
  float4 f0 = *(const float4*)(src + i);
  float4 f1 = *(const float4*)(src + i + 4);
  bf16x8 v;
  v[0] = (__bf16)f0.x; v[1] = (__bf16)f0.y;
  v[2] = (__bf16)f0.z; v[3] = (__bf16)f0.w;
  v[4] = (__bf16)f1.x; v[5] = (__bf16)f1.y;
  v[6] = (__bf16)f1.z; v[7] = (__bf16)f1.w;
  *(bf16x8*)(dst + i) = v;
}

// ---------------------------------------------------------------------------
// Kernel 1: generic NT GEMM, C[m,n] = sum_k A[m,k] * Bw[n,k], K = 2048 (bf16).
// Block: 128 threads (4 waves). Block tile 64x64; wave tile 16x64 (4 accums).
// A/B tiles staged 128-deep in LDS via GLOBAL_LOAD_ASYNC_TO_LDS_B128.
// Fragment loads are batched per depth chunk so DS waits are partial and
// LDS reads overlap the XDL pipeline.
// ---------------------------------------------------------------------------
__global__ __launch_bounds__(128)
void gemm_nt_bf16_kernel(const __bf16* __restrict__ A,   // [M][2048]
                         const __bf16* __restrict__ Bw,  // [N][2048]
                         float* __restrict__ C,          // [M][ldc]
                         int ldc) {
  __shared__ __bf16 As[64][128];   // 16 KB
  __shared__ __bf16 Bs[64][128];   // 16 KB

  const int n0 = blockIdx.x * 64;
  const int m0 = blockIdx.y * 64;

  const int tid  = threadIdx.x;
  const int lane = tid & 31;
  const int wave = tid >> 5;        // 16-row M strip
  const int hf   = (lane >> 4) & 1;
  const int l16  = lane & 15;

  const int srow = tid >> 1;            // staging row 0..63
  const int scol = (tid & 1) * 64;      // staging col base 0/64

  const __bf16* aptr = A  + (size_t)(m0 + srow) * HID + scol;
  const __bf16* bptr = Bw + (size_t)(n0 + srow) * HID + scol;

  f32x8 acc[4];
#pragma unroll
  for (int t = 0; t < 4; ++t)
    acc[t] = (f32x8){0.f, 0.f, 0.f, 0.f, 0.f, 0.f, 0.f, 0.f};

  for (int k0 = 0; k0 < HID; k0 += 128) {
#if HAVE_ASYNC_LDS
    // Async DMA the 64x128 A and B bf16 panels into LDS (16 B / lane / op).
#pragma unroll
    for (int j = 0; j < 64; j += 8) {
      async_copy16(aptr + k0 + j, &As[srow][scol + j]);
      async_copy16(bptr + k0 + j, &Bs[srow][scol + j]);
    }
    async_wait0();
#else
#pragma unroll
    for (int j = 0; j < 64; j += 8) {
      *(bf16x8*)&As[srow][scol + j] = *(const bf16x8*)(aptr + k0 + j);
      *(bf16x8*)&Bs[srow][scol + j] = *(const bf16x8*)(bptr + k0 + j);
    }
#endif
    if (k0 + 128 < HID) {
      __builtin_prefetch(aptr + k0 + 128, 0, 1);   // global_prefetch_b8
      __builtin_prefetch(bptr + k0 + 128, 0, 1);
    }
    __syncthreads();

#pragma unroll
    for (int c = 0; c < 4; ++c) {       // depth chunks of 32
      // A 16x32 bf16 lane layout: row = lane%16; K = half*16 + e (contig)
      AFrag af;
      af.h8[0] = *(const bf16x8*)&As[wave * 16 + l16][c * 32 + hf * 16];
      af.h8[1] = *(const bf16x8*)&As[wave * 16 + l16][c * 32 + hf * 16 + 8];
      // Batch all 4 B fragments first so the WMMA chain only needs
      // partial s_wait_dscnt values (DS completes in order per wave).
      AFrag bfg[4];
#pragma unroll
      for (int t = 0; t < 4; ++t) {
        // B 32x16 bf16 lane layout: n = lane%16; K = half*16 + e (contig)
        bfg[t].v = *(const bf16x16*)&Bs[t * 16 + l16][c * 32 + hf * 16];
      }
#pragma unroll
      for (int t = 0; t < 4; ++t) {
        acc[t] = wmma_bf16(af.v, bfg[t].v, acc[t]);
      }
    }
    __syncthreads();
  }

  // C layout: VGPR i -> row = half*8 + i, col = lane%16
#pragma unroll
  for (int t = 0; t < 4; ++t) {
#pragma unroll
    for (int i = 0; i < 8; ++i) {
      C[(size_t)(m0 + wave * 16 + hf * 8 + i) * ldc + n0 + t * 16 + l16] =
          acc[t][i];
    }
  }
}

// ---------------------------------------------------------------------------
// Kernel 2: RoPE + bf16 conversion + relayout.
//   Q  -> [b, h,  s, d]   bf16   (rope)
//   K  -> [b, kh, s, d]   bf16   (rope)
//   V  -> [b, kh, d, s]   bf16   (transposed so PV B-fragments are contiguous)
// ---------------------------------------------------------------------------
__global__ __launch_bounds__(256)
void rope_cvt_kernel(const float* __restrict__ qkv,
                     const int* __restrict__ pos_ids,
                     __bf16* __restrict__ Qbf,
                     __bf16* __restrict__ Kbf,
                     __bf16* __restrict__ Vt) {
  const unsigned idx = blockIdx.x * 256u + threadIdx.x;   // < MTOT*NTOT
  const int n = (int)(idx % NTOT);
  const unsigned m = idx / NTOT;
  const int b = (int)(m / S_DIM);
  const int s = (int)(m % S_DIM);
  const float v = qkv[(size_t)m * NTOT + n];

  if (n < NQ + NKV) {               // Q or K: apply rope
    const int d  = n & (HD - 1);
    const int dh = d & 63;
    const int pos = pos_ids[b * S_DIM + s];
    // inv_freq = theta^(-2*dh/128) = exp(-dh * ln(10000)/64)
    const float invf = __expf(-(float)dh * (9.210340371976184f / 64.0f));
    const float f = (float)pos * invf;
    float sn, cs;
    __sincosf(f, &sn, &cs);
    const float other = (d < 64) ? qkv[(size_t)m * NTOT + n + 64]
                                 : qkv[(size_t)m * NTOT + n - 64];
    const float r = (d < 64) ? (v * cs - other * sn) : (v * cs + other * sn);
    if (n < NQ) {
      const int h = n / HD;
      Qbf[(((size_t)b * NH + h) * S_DIM + s) * HD + d] = (__bf16)r;
    } else {
      const int kh = (n - NQ) / HD;
      Kbf[(((size_t)b * NKVH + kh) * S_DIM + s) * HD + d] = (__bf16)r;
    }
  } else {                          // V: transpose to [b, kh, d, s]
    const int nn = n - NQ - NKV;
    const int kh = nn / HD;
    const int d  = nn & (HD - 1);
    Vt[(((size_t)b * NKVH + kh) * HD + d) * S_DIM + s] = (__bf16)v;
  }
}

// ---------------------------------------------------------------------------
// Kernel 3: causal flash attention, GQA 4:1.
// Grid: (S/64, H, B). Block: 128 threads = 4 waves; wave w owns Q rows
// [q0+16w, q0+16w+16). Online softmax in f32; P re-laid out via per-wave LDS.
// ---------------------------------------------------------------------------
__global__ __launch_bounds__(128)
void flash_attn_kernel(const __bf16* __restrict__ Qbf,
                       const __bf16* __restrict__ Kbf,
                       const __bf16* __restrict__ Vt,
                       __bf16* __restrict__ aout) {
  const int qt = blockIdx.x;            // q-tile (64 rows)
  const int h  = blockIdx.y;
  const int b  = blockIdx.z;
  const int kh = h / (NH / NKVH);

  const int tid  = threadIdx.x;
  const int lane = tid & 31;
  const int wave = tid >> 5;
  const int hf   = (lane >> 4) & 1;
  const int l16  = lane & 15;

  const int q0   = qt * 64;
  const int qrow = q0 + wave * 16 + l16;     // A-fragment row (lane%16)

  __shared__ __bf16 Pst[4][16][64];          // per-wave P strip, 8 KB

  const __bf16* Qb = Qbf + ((size_t)(b * NH + h) * S_DIM) * HD;
  const __bf16* Kb = Kbf + ((size_t)(b * NKVH + kh) * S_DIM) * HD;
  const __bf16* Vb = Vt  + ((size_t)(b * NKVH + kh) * HD) * S_DIM;

  // Q A-fragments for the 4 depth chunks (contraction d = c*32 .. c*32+31)
  AFrag qf[4];
#pragma unroll
  for (int c = 0; c < 4; ++c) {
    const __bf16* r = Qb + (size_t)qrow * HD + c * 32;
    qf[c].h8[0] = *(const bf16x8*)(r + hf * 16);
    qf[c].h8[1] = *(const bf16x8*)(r + hf * 16 + 8);
  }

  f32x8 o[8];
  float mrow[8], lrow[8];
#pragma unroll
  for (int i = 0; i < 8; ++i) {
    o[i] = (f32x8){0.f, 0.f, 0.f, 0.f, 0.f, 0.f, 0.f, 0.f};
    mrow[i] = -3.0e38f;
    lrow[i] = 0.f;
  }

  for (int j = 0; j <= qt; ++j) {            // causal: k-tiles up to q-tile
    const int k0 = j * 64;
    if (j < qt) {                            // prefetch next K tile rows
      __builtin_prefetch(Kb + (size_t)(k0 + 64 + (lane << 1)) * HD, 0, 1);
    }

    // ---- scores S = Q K^T, chunk-major with batched K fragments ----
    f32x8 sacc[4];
#pragma unroll
    for (int t = 0; t < 4; ++t)
      sacc[t] = (f32x8){0.f, 0.f, 0.f, 0.f, 0.f, 0.f, 0.f, 0.f};
#pragma unroll
    for (int c = 0; c < 4; ++c) {
      AFrag kf[4];
#pragma unroll
      for (int t = 0; t < 4; ++t) {
        // K^T B-fragment: n = key = k0+t*16+lane%16, contr d = c*32+half*16+e
        kf[t].v = *(const bf16x16*)(Kb + (size_t)(k0 + t * 16 + l16) * HD +
                                    c * 32 + hf * 16);
      }
#pragma unroll
      for (int t = 0; t < 4; ++t) {
        sacc[t] = wmma_bf16(qf[c].v, kf[t].v, sacc[t]);
      }
    }

    // ---- scale + causal mask ----
#pragma unroll
    for (int t = 0; t < 4; ++t) {
      const int key = k0 + t * 16 + l16;
#pragma unroll
      for (int i = 0; i < 8; ++i) {
        const int row = q0 + wave * 16 + hf * 8 + i;
        const float sv = sacc[t][i] * SCALE_F;
        sacc[t][i] = (key > row) ? NEG_F : sv;
      }
    }

    // ---- online softmax (row = one VGPR slot across 16 lanes) ----
#pragma unroll
    for (int i = 0; i < 8; ++i) {
      float vmax = sacc[0][i];
      vmax = fmaxf(vmax, sacc[1][i]);
      vmax = fmaxf(vmax, sacc[2][i]);
      vmax = fmaxf(vmax, sacc[3][i]);
      vmax = rowmax16(vmax);
      const float mnew  = fmaxf(mrow[i], vmax);
      const float alpha = __expf(mrow[i] - mnew);
#pragma unroll
      for (int dn = 0; dn < 8; ++dn) o[dn][i] *= alpha;
      float psum = 0.f;
#pragma unroll
      for (int t = 0; t < 4; ++t) {
        const float p = __expf(sacc[t][i] - mnew);
        psum += p;
        Pst[wave][hf * 8 + i][t * 16 + l16] = (__bf16)p;
      }
      lrow[i] = lrow[i] * alpha + rowsum16(psum);
      mrow[i] = mnew;
    }

    // ---- O += P V  (P via LDS -> A-fragments; V pre-transposed) ----
#pragma unroll
    for (int c = 0; c < 2; ++c) {            // key chunks of 32
      AFrag pf;
      pf.h8[0] = *(const bf16x8*)&Pst[wave][l16][c * 32 + hf * 16];
      pf.h8[1] = *(const bf16x8*)&Pst[wave][l16][c * 32 + hf * 16 + 8];
#pragma unroll
      for (int g = 0; g < 2; ++g) {          // d sub-tiles in groups of 4
        AFrag vf[4];
#pragma unroll
        for (int q = 0; q < 4; ++q) {
          const int dn = g * 4 + q;
          // B-fragment: n = d = dn*16+lane%16, contr key = k0+c*32+half*16+e
          vf[q].v = *(const bf16x16*)(Vb + (size_t)(dn * 16 + l16) * S_DIM +
                                      k0 + c * 32 + hf * 16);
        }
#pragma unroll
        for (int q = 0; q < 4; ++q) {
          o[g * 4 + q] = wmma_bf16(pf.v, vf[q].v, o[g * 4 + q]);
        }
      }
    }
  }

  // ---- normalize and store attn output [b, s, h*D + d] (bf16) ----
#pragma unroll
  for (int i = 0; i < 8; ++i) {
    const float inv = 1.0f / lrow[i];
    const int s = q0 + wave * 16 + hf * 8 + i;
    __bf16* dst = aout + ((size_t)b * S_DIM + s) * NQ + h * HD;
#pragma unroll
    for (int dn = 0; dn < 8; ++dn) {
      dst[dn * 16 + l16] = (__bf16)(o[dn][i] * inv);
    }
  }
}

// ---------------------------------------------------------------------------
// Host-side launch. Workspace layout (bytes):
//   [0,          16777216)   x    bf16  [4096][2048]
//   [16777216,   29360128)   Wqkv bf16  [3072][2048]  (Wq | Wk | Wv rows)
//   [29360128,   37748736)   Wo   bf16  [2048][2048]
//   [37748736,   88080384)   qkv  fp32  [4096][3072]
//   [88080384,  104857600)   Q    bf16  [2][16][2048][128]
//   [104857600, 109051904)   K    bf16  [2][4][2048][128]
//   [109051904, 113246208)   V^T  bf16  [2][4][128][2048]
//   [113246208, 130023424)   attn bf16  [4096][2048]
// ---------------------------------------------------------------------------
extern "C" void kernel_launch(void* const* d_in, const int* in_sizes, int n_in,
                              void* d_out, int out_size, void* d_ws,
                              size_t ws_size, hipStream_t stream) {
  const float* x       = (const float*)d_in[0];
  // d_in[1]: attn_mask (exact causal, reproduced analytically in-kernel)
  const int*   pos_ids = (const int*)d_in[2];
  const float* Wq      = (const float*)d_in[3];
  const float* Wk      = (const float*)d_in[4];
  const float* Wv      = (const float*)d_in[5];
  const float* Wo      = (const float*)d_in[6];
  float*       out     = (float*)d_out;

  char* ws = (char*)d_ws;
  __bf16* xbf    = (__bf16*)(ws);
  __bf16* Wqkvbf = (__bf16*)(ws + 16777216);
  __bf16* Wobf   = (__bf16*)(ws + 29360128);
  float*  qkv    = (float*)(ws + 37748736);
  __bf16* Qbf    = (__bf16*)(ws + 88080384);
  __bf16* Kbf    = (__bf16*)(ws + 104857600);
  __bf16* Vt     = (__bf16*)(ws + 109051904);
  __bf16* aout   = (__bf16*)(ws + 113246208);

  // 0) fp32 -> bf16 pre-conversion (x, Wq|Wk|Wv packed, Wo)
  cvt_bf16_kernel<<<dim3(4096), 256, 0, stream>>>(x, xbf);                   // 4096*2048
  cvt_bf16_kernel<<<dim3(2048), 256, 0, stream>>>(Wq, Wqkvbf);               // 2048*2048
  cvt_bf16_kernel<<<dim3(512),  256, 0, stream>>>(Wk, Wqkvbf + NQ * HID);    //  512*2048
  cvt_bf16_kernel<<<dim3(512),  256, 0, stream>>>(Wv, Wqkvbf + (NQ + NKV) * HID);
  cvt_bf16_kernel<<<dim3(2048), 256, 0, stream>>>(Wo, Wobf);                 // 2048*2048

  // 1) fused QKV projection: one NT GEMM, N = 3072. grid = (3072/64, 4096/64)
  gemm_nt_bf16_kernel<<<dim3(NTOT / 64, MTOT / 64), 128, 0, stream>>>(
      xbf, Wqkvbf, qkv, NTOT);

  // 2) RoPE + convert + relayout: 4096*3072 elements
  rope_cvt_kernel<<<dim3((MTOT * NTOT) / 256), 256, 0, stream>>>(
      qkv, pos_ids, Qbf, Kbf, Vt);

  // 3) Flash attention: grid = (S/64, H, B)
  flash_attn_kernel<<<dim3(S_DIM / 64, NH, B_DIM), 128, 0, stream>>>(
      Qbf, Kbf, Vt, aout);

  // 4) Output projection: grid = (2048/64, 4096/64)
  gemm_nt_bf16_kernel<<<dim3(HID / 64, MTOT / 64), 128, 0, stream>>>(
      aout, Wobf, out, HID);
}
